// CKAMap_13838384628035
// MI455X (gfx1250) — compile-verified, matrix operand order checked by reading, not run on hardware
//
#include <hip/hip_runtime.h>
#include <math.h>

typedef __attribute__((ext_vector_type(16))) __bf16 v16bf;
typedef __attribute__((ext_vector_type(8)))  __bf16 v8bf;
typedef __attribute__((ext_vector_type(8)))  float  v8f;

#define L_LAYERS 24
#define NB 2048
#define ND 2048
#define NBLK_RED 512

// ---------------------------------------------------------------------------
// Kernel 0: zero the double-precision pair partials (+ diag partials)
// ---------------------------------------------------------------------------
__global__ void zero_partial_kernel(double* p, int n) {
    int i = blockIdx.x * blockDim.x + threadIdx.x;
    if (i < n) p[i] = 0.0;
}

// ---------------------------------------------------------------------------
// Kernel 1: column means over batch: mean[l][d] = (1/B) sum_b X[l][b][d]
// grid (24, 8), block 256
// ---------------------------------------------------------------------------
__global__ void means_kernel(const float* __restrict__ X, float* __restrict__ mean) {
    int l = blockIdx.x;
    int d = blockIdx.y * 256 + threadIdx.x;
    const float* p = X + (size_t)l * NB * ND + d;
    float s = 0.f;
    #pragma unroll 8
    for (int b = 0; b < NB; ++b) s += p[(size_t)b * ND];
    mean[l * ND + d] = s * (1.0f / (float)NB);
}

// ---------------------------------------------------------------------------
// Kernel 1b (fast path): precompute centered split-bf16 arrays.
// x = X - mean ; Xhi = bf16(x) ; Xlo = bf16(x - float(Xhi))
// Each thread owns a pair of d-columns; packed 4B stores.
// grid (24, ND/512), block 256
// ---------------------------------------------------------------------------
union BF2 { __bf16 h[2]; unsigned u; };

__global__ void split_kernel(const float* __restrict__ X,
                             const float* __restrict__ mean,
                             __bf16* __restrict__ Xhi, __bf16* __restrict__ Xlo) {
    int l = blockIdx.x;
    int d = (blockIdx.y * 256 + threadIdx.x) * 2;
    float2 m2 = *(const float2*)(mean + l * ND + d);
    const float* src = X + (size_t)l * NB * ND + d;
    __bf16* ph = Xhi + (size_t)l * NB * ND + d;
    __bf16* pl = Xlo + (size_t)l * NB * ND + d;
    for (int b = 0; b < NB; ++b) {
        float2 x2 = *(const float2*)(src + (size_t)b * ND);
        float x0 = x2.x - m2.x, x1 = x2.y - m2.y;
        BF2 hh, ll;
        hh.h[0] = (__bf16)x0;
        hh.h[1] = (__bf16)x1;
        ll.h[0] = (__bf16)(x0 - (float)hh.h[0]);
        ll.h[1] = (__bf16)(x1 - (float)hh.h[1]);
        *(unsigned*)(ph + (size_t)b * ND) = hh.u;
        *(unsigned*)(pl + (size_t)b * ND) = ll.u;
    }
}

// ---------------------------------------------------------------------------
// Operand load: v16bf elements 0..7 = 8 consecutive K at ka,
//               elements 8..15     = 8 consecutive K at ka+16.
// ---------------------------------------------------------------------------
__device__ __forceinline__ v16bf load_op(const __bf16* p) {
    v8bf a = *(const v8bf*)(p);
    v8bf b = *(const v8bf*)(p + 16);
    return __builtin_shufflevector(a, b, 0, 1, 2, 3, 4, 5, 6, 7,
                                   8, 9, 10, 11, 12, 13, 14, 15);
}

// ---------------------------------------------------------------------------
// Shared epilogue: store acc with symmetry mask (keep col >= global row,
// zero below diagonal).  C/D layout: VGPR g -> M = g + 8*half, N = lane%16.
// ---------------------------------------------------------------------------
__device__ __forceinline__ void store_tiles(float* __restrict__ Gbuf, v8f acc[4][4],
                                            int l, int chunk, int m0, int n0,
                                            int b0, int half, int r) {
    #pragma unroll
    for (int mt = 0; mt < 4; ++mt)
        #pragma unroll
        for (int g = 0; g < 8; ++g) {
            int lrow = m0 + mt * 16 + g + 8 * half;
            int growr = b0 + lrow;
            size_t row = (size_t)l * chunk + (size_t)lrow;
            float* dst = Gbuf + row * ND + n0 + r;
            #pragma unroll
            for (int nt = 0; nt < 4; ++nt) {
                int col = n0 + nt * 16 + r;
                dst[nt * 16] = (col >= growr) ? acc[mt][nt][g] : 0.f;
            }
        }
}

__device__ __forceinline__ void store_zeros(float* __restrict__ Gbuf,
                                            int l, int chunk, int m0, int n0,
                                            int half, int r) {
    #pragma unroll
    for (int mt = 0; mt < 4; ++mt)
        #pragma unroll
        for (int g = 0; g < 8; ++g) {
            size_t row = (size_t)l * chunk + (size_t)(m0 + mt * 16 + g + 8 * half);
            float* dst = Gbuf + row * ND + n0 + r;
            #pragma unroll
            for (int nt = 0; nt < 4; ++nt) dst[nt * 16] = 0.f;
        }
}

// ---------------------------------------------------------------------------
// Kernel 2 (fast): Gram rows from precomputed split-bf16 operands.
// grid (24, chunk/64, 8) ; block 128 (4 waves); wave tile 64x64, WG 64x256.
// Upper-triangle only (symmetry); sub-diagonal WGs emit zeros.
// ---------------------------------------------------------------------------
__global__ __launch_bounds__(128, 1)
void gram_fast_kernel(const __bf16* __restrict__ Xhi, const __bf16* __restrict__ Xlo,
                      float* __restrict__ Gbuf, int b0, int chunk) {
    const int l    = blockIdx.x;
    const int m0   = blockIdx.y * 64;
    const int lane = threadIdx.x & 31;
    const int wave = threadIdx.x >> 5;
    const int half = lane >> 4;
    const int r    = lane & 15;
    const int n0   = blockIdx.z * 256 + wave * 64;

    if (blockIdx.z * 256 + 256 <= b0 + m0) {          // tile fully below diagonal
        store_zeros(Gbuf, l, chunk, m0, n0, half, r);
        return;
    }

    const __bf16* Hl = Xhi + (size_t)l * NB * ND;
    const __bf16* Ll = Xlo + (size_t)l * NB * ND;

    size_t aoff[4], boff[4];
    #pragma unroll
    for (int t = 0; t < 4; ++t) {
        aoff[t] = (size_t)(b0 + m0 + t * 16 + r) * ND;
        boff[t] = (size_t)(n0 + t * 16 + r) * ND;
    }

    v8f acc[4][4];
    #pragma unroll
    for (int mt = 0; mt < 4; ++mt)
        #pragma unroll
        for (int nt = 0; nt < 4; ++nt) acc[mt][nt] = (v8f)(0.0f);

    for (int k0 = 0; k0 < ND; k0 += 32) {
        const int ka = k0 + half * 8;
        v16bf ahi[4], alo[4];
        #pragma unroll
        for (int t = 0; t < 4; ++t) {
            ahi[t] = load_op(Hl + aoff[t] + ka);
            alo[t] = load_op(Ll + aoff[t] + ka);
        }
        #pragma unroll
        for (int nt = 0; nt < 4; ++nt) {
            v16bf bhi = load_op(Hl + boff[nt] + ka);
            v16bf blo = load_op(Ll + boff[nt] + ka);
            #pragma unroll
            for (int mt = 0; mt < 4; ++mt) {
                acc[mt][nt] = __builtin_amdgcn_wmma_f32_16x16x32_bf16(
                    false, ahi[mt], false, bhi, (short)0, acc[mt][nt], false, false);
                acc[mt][nt] = __builtin_amdgcn_wmma_f32_16x16x32_bf16(
                    false, ahi[mt], false, blo, (short)0, acc[mt][nt], false, false);
                acc[mt][nt] = __builtin_amdgcn_wmma_f32_16x16x32_bf16(
                    false, alo[mt], false, bhi, (short)0, acc[mt][nt], false, false);
            }
        }
    }

    store_tiles(Gbuf, acc, l, chunk, m0, n0, b0, half, r);
}

// ---------------------------------------------------------------------------
// Kernel 2 (fallback): convert fp32 -> split bf16 inside the GEMM loop.
// Same tiling / symmetry handling as the fast kernel.
// ---------------------------------------------------------------------------
__global__ __launch_bounds__(128, 1)
void gram_slow_kernel(const float* __restrict__ X, const float* __restrict__ mean,
                      float* __restrict__ Gbuf, int b0, int chunk) {
    const int l    = blockIdx.x;
    const int m0   = blockIdx.y * 64;
    const int lane = threadIdx.x & 31;
    const int wave = threadIdx.x >> 5;
    const int half = lane >> 4;
    const int r    = lane & 15;
    const int n0   = blockIdx.z * 256 + wave * 64;

    if (blockIdx.z * 256 + 256 <= b0 + m0) {
        store_zeros(Gbuf, l, chunk, m0, n0, half, r);
        return;
    }

    __shared__ float smean[ND];
    const float* Xl = X + (size_t)l * NB * ND;
    for (int i = threadIdx.x; i < ND; i += 128) smean[i] = mean[l * ND + i];
    __syncthreads();

    int arow[4], brow[4];
    #pragma unroll
    for (int t = 0; t < 4; ++t) {
        arow[t] = b0 + m0 + t * 16 + r;
        brow[t] = n0 + t * 16 + r;
    }

    v8f acc[4][4];
    #pragma unroll
    for (int mt = 0; mt < 4; ++mt)
        #pragma unroll
        for (int nt = 0; nt < 4; ++nt) acc[mt][nt] = (v8f)(0.0f);

    for (int k0 = 0; k0 < ND; k0 += 32) {
        const int ka = k0 + half * 8;
        v16bf ahi[4], alo[4];
        #pragma unroll
        for (int t = 0; t < 4; ++t) {
            const float* base = Xl + (size_t)arow[t] * ND + ka;
            float4 f0 = *(const float4*)(base);
            float4 f1 = *(const float4*)(base + 4);
            float4 f2 = *(const float4*)(base + 16);
            float4 f3 = *(const float4*)(base + 20);
            float v[16] = {f0.x, f0.y, f0.z, f0.w, f1.x, f1.y, f1.z, f1.w,
                           f2.x, f2.y, f2.z, f2.w, f3.x, f3.y, f3.z, f3.w};
            #pragma unroll
            for (int e = 0; e < 16; ++e) {
                float m = (e < 8) ? smean[ka + e] : smean[ka + 8 + e];
                float x = v[e] - m;
                __bf16 h = (__bf16)x;
                ahi[t][e] = h;
                alo[t][e] = (__bf16)(x - (float)h);
            }
        }
        #pragma unroll
        for (int nt = 0; nt < 4; ++nt) {
            v16bf bhi, blo;
            {
                const float* base = Xl + (size_t)brow[nt] * ND + ka;
                float4 f0 = *(const float4*)(base);
                float4 f1 = *(const float4*)(base + 4);
                float4 f2 = *(const float4*)(base + 16);
                float4 f3 = *(const float4*)(base + 20);
                float v[16] = {f0.x, f0.y, f0.z, f0.w, f1.x, f1.y, f1.z, f1.w,
                               f2.x, f2.y, f2.z, f2.w, f3.x, f3.y, f3.z, f3.w};
                #pragma unroll
                for (int e = 0; e < 16; ++e) {
                    float m = (e < 8) ? smean[ka + e] : smean[ka + 8 + e];
                    float x = v[e] - m;
                    __bf16 h = (__bf16)x;
                    bhi[e] = h;
                    blo[e] = (__bf16)(x - (float)h);
                }
            }
            #pragma unroll
            for (int mt = 0; mt < 4; ++mt) {
                acc[mt][nt] = __builtin_amdgcn_wmma_f32_16x16x32_bf16(
                    false, ahi[mt], false, bhi, (short)0, acc[mt][nt], false, false);
                acc[mt][nt] = __builtin_amdgcn_wmma_f32_16x16x32_bf16(
                    false, ahi[mt], false, blo, (short)0, acc[mt][nt], false, false);
                acc[mt][nt] = __builtin_amdgcn_wmma_f32_16x16x32_bf16(
                    false, alo[mt], false, bhi, (short)0, acc[mt][nt], false, false);
            }
        }
    }

    store_tiles(Gbuf, acc, l, chunk, m0, n0, b0, half, r);
}

// ---------------------------------------------------------------------------
// Kernel 3: pair reduction over one chunk (Gbuf holds upper-triangle values,
// zeros below diagonal). partial[block][t] += in fixed order (deterministic).
// ---------------------------------------------------------------------------
__global__ __launch_bounds__(576)
void reduce_kernel(const float* __restrict__ Gbuf, double* __restrict__ partial,
                   int chunk) {
    __shared__ float tile[256 * 32];
    const size_t P = (size_t)chunk * ND;
    const int t = threadIdx.x;
    const int i = t / 24, j = t % 24;

    double acc = 0.0;
    for (size_t base = (size_t)blockIdx.x * 256; base < P;
         base += (size_t)gridDim.x * 256) {
        for (int idx = t; idx < 24 * 256; idx += 576) {
            int l = idx >> 8, p = idx & 255;
            tile[p * 32 + l] = Gbuf[(size_t)l * P + base + p];
        }
        __syncthreads();
        float a = 0.f;
        #pragma unroll 8
        for (int p = 0; p < 256; ++p)
            a += tile[p * 32 + i] * tile[p * 32 + j];
        acc += (double)a;
        __syncthreads();
    }
    partial[(size_t)blockIdx.x * 576 + t] += acc;
}

// ---------------------------------------------------------------------------
// Kernel 3b: diagonal products (needed for hsic = 2*S_upper - S_diag).
// Single block, fixed-order accumulation across chunks (deterministic).
// ---------------------------------------------------------------------------
__global__ __launch_bounds__(576)
void diag_kernel(const float* __restrict__ Gbuf, double* __restrict__ pdiag,
                 int b0, int chunk) {
    __shared__ float vals[64][25];
    const size_t P = (size_t)chunk * ND;
    const int t = threadIdx.x;
    const int i = t / 24, j = t % 24;
    double acc = 0.0;
    for (int r0 = 0; r0 < chunk; r0 += 64) {
        for (int idx = t; idx < 64 * 24; idx += 576) {
            int rr = idx / 24, l = idx % 24;
            vals[rr][l] = Gbuf[(size_t)l * P + (size_t)(r0 + rr) * ND + (b0 + r0 + rr)];
        }
        __syncthreads();
        float a = 0.f;
        #pragma unroll 8
        for (int rr = 0; rr < 64; ++rr) a += vals[rr][i] * vals[rr][j];
        acc += (double)a;
        __syncthreads();
    }
    pdiag[t] += acc;
}

// ---------------------------------------------------------------------------
// Kernel 4: final sum + symmetry correction + CKA normalization.
// ---------------------------------------------------------------------------
__global__ __launch_bounds__(576)
void final_kernel(const double* __restrict__ partial, const double* __restrict__ pdiag,
                  int nblk, float* __restrict__ out) {
    __shared__ double hs[576];
    const int t = threadIdx.x;
    double s = 0.0;
    for (int b = 0; b < nblk; ++b) s += partial[(size_t)b * 576 + t];
    hs[t] = 2.0 * s - pdiag[t];          // full = 2*upper(incl diag) - diag
    __syncthreads();
    const int i = t / 24, j = t % 24;
    double denom = sqrt(hs[i * 24 + i] * hs[j * 24 + j]);
    out[t] = (float)(hs[t] / denom);
}

// ---------------------------------------------------------------------------
extern "C" void kernel_launch(void* const* d_in, const int* in_sizes, int n_in,
                              void* d_out, int out_size, void* d_ws, size_t ws_size,
                              hipStream_t stream) {
    const float* X = (const float*)d_in[0];
    float* out = (float*)d_out;

    // fixed workspace layout
    float* mean = (float*)d_ws;
    size_t off = (size_t)L_LAYERS * ND * sizeof(float);                 // 196608
    double* partial = (double*)((char*)d_ws + off);
    off += (size_t)NBLK_RED * 576 * sizeof(double);                     // +2359296
    double* pdiag = (double*)((char*)d_ws + off);
    off += 576 * sizeof(double);                                        // +4608

    const size_t elems = (size_t)L_LAYERS * NB * ND;
    const size_t split_bytes = 2 * elems * sizeof(__bf16);              // 402 MB
    const size_t min_gbuf = (size_t)L_LAYERS * 64 * ND * sizeof(float); // 12.6 MB

    bool fast = (ws_size >= off + split_bytes + min_gbuf);

    __bf16* Xhi = nullptr;
    __bf16* Xlo = nullptr;
    if (fast) {
        Xhi = (__bf16*)((char*)d_ws + off);
        Xlo = Xhi + elems;
        off += split_bytes;
    }
    float* Gbuf = (float*)((char*)d_ws + off);
    size_t avail = (ws_size > off) ? ws_size - off : 0;

    int chunk = 64;
    const int cand[6] = {2048, 1024, 512, 256, 128, 64};
    for (int c = 0; c < 6; ++c) {
        if ((size_t)L_LAYERS * cand[c] * ND * sizeof(float) <= avail) {
            chunk = cand[c];
            break;
        }
    }

    zero_partial_kernel<<<(NBLK_RED * 576 + 576 + 255) / 256, 256, 0, stream>>>(
        partial, NBLK_RED * 576 + 576);
    means_kernel<<<dim3(L_LAYERS, ND / 256), 256, 0, stream>>>(X, mean);
    if (fast)
        split_kernel<<<dim3(L_LAYERS, ND / 512), 256, 0, stream>>>(X, mean, Xhi, Xlo);

    for (int b0 = 0; b0 < NB; b0 += chunk) {
        if (fast)
            gram_fast_kernel<<<dim3(L_LAYERS, chunk / 64, ND / 256), 128, 0, stream>>>(
                Xhi, Xlo, Gbuf, b0, chunk);
        else
            gram_slow_kernel<<<dim3(L_LAYERS, chunk / 64, ND / 256), 128, 0, stream>>>(
                X, mean, Gbuf, b0, chunk);
        reduce_kernel<<<NBLK_RED, 576, 0, stream>>>(Gbuf, partial, chunk);
        diag_kernel<<<1, 576, 0, stream>>>(Gbuf, pdiag, b0, chunk);
    }

    final_kernel<<<1, 576, 0, stream>>>(partial, pdiag, NBLK_RED, out);
}
